// SAGELayer_14697378087214
// MI455X (gfx1250) — matrix operand order repaired and checked by reading, not compiled, not run on hardware
//
#include <hip/hip_runtime.h>
#include <hip/hip_bf16.h>

typedef __attribute__((ext_vector_type(16))) __bf16 v16bf;
typedef __attribute__((ext_vector_type(8)))  __bf16 v8bf;
typedef __attribute__((ext_vector_type(8)))  float  v8f;

#define DDIM 512

// ---------------------------------------------------------------------------
// Kernel 1: zero the scatter accumulators (ws is poisoned; must re-zero per call)
// ---------------------------------------------------------------------------
__global__ void sage_zero(float* __restrict__ agg, float* __restrict__ deg,
                          int nagg, int ndeg) {
    int i = blockIdx.x * blockDim.x + threadIdx.x;
    int total = nagg + ndeg;
    if (i < total) {
        if (i < nagg) agg[i] = 0.0f;
        else          deg[i - nagg] = 0.0f;
    }
}

// ---------------------------------------------------------------------------
// Kernel 2: transpose + convert weights to bf16: WT[n][k] = (bf16)W[k][n]
// ---------------------------------------------------------------------------
__global__ void sage_wt(const float* __restrict__ Wl, const float* __restrict__ Wr,
                        __bf16* __restrict__ WTl, __bf16* __restrict__ WTr) {
    int i = blockIdx.x * blockDim.x + threadIdx.x;   // i = k*512 + n (row-major W)
    if (i >= DDIM * DDIM) return;
    int n = i & (DDIM - 1);
    int k = i >> 9;
    WTl[n * DDIM + k] = (__bf16)Wl[i];
    WTr[n * DDIM + k] = (__bf16)Wr[i];
}

// ---------------------------------------------------------------------------
// Kernel 3: x -> bf16 (independent of scatter)
// ---------------------------------------------------------------------------
__global__ void sage_xcvt(const float* __restrict__ x, __bf16* __restrict__ xb,
                          int total) {
    int i = (blockIdx.x * blockDim.x + threadIdx.x) * 4;
    if (i >= total) return;
    const float4 v = *(const float4*)(x + i);
    v8bf o4 = {};  // only first 4 used; write as 8-byte chunk
    __bf16* o = xb + i;
    o[0] = (__bf16)v.x; o[1] = (__bf16)v.y; o[2] = (__bf16)v.z; o[3] = (__bf16)v.w;
    (void)o4;
}

// ---------------------------------------------------------------------------
// Kernel 4: edge scatter-add (mean-aggregation numerator + degree)
// One block per edge, 128 threads x float4 = 512 channels.
// ---------------------------------------------------------------------------
__global__ __launch_bounds__(128)
void sage_scatter(const float* __restrict__ x, const int* __restrict__ src,
                  const int* __restrict__ dst, float* __restrict__ agg,
                  float* __restrict__ deg, int E) {
    int e = blockIdx.x;
    if (e >= E) return;
    int s = src[e];
    int d = dst[e];
    int c = threadIdx.x << 2;                          // 4 channels / thread
    const float4 v = *(const float4*)(x + (size_t)s * DDIM + c);
    float* a = agg + (size_t)d * DDIM + c;
    atomicAdd(a + 0, v.x);
    atomicAdd(a + 1, v.y);
    atomicAdd(a + 2, v.z);
    atomicAdd(a + 3, v.w);
    if (threadIdx.x == 0) atomicAdd(deg + d, 1.0f);
}

// ---------------------------------------------------------------------------
// Kernel 5: mean + bf16 convert: meanb[m][c] = bf16(agg[m][c] / max(deg[m],1))
// ---------------------------------------------------------------------------
__global__ void sage_mean(const float* __restrict__ agg, const float* __restrict__ deg,
                          __bf16* __restrict__ meanb, int total) {
    int i = (blockIdx.x * blockDim.x + threadIdx.x) * 4;
    if (i >= total) return;
    int m = i >> 9;
    float r = 1.0f / fmaxf(deg[m], 1.0f);
    const float4 v = *(const float4*)(agg + i);
    __bf16* o = meanb + i;
    o[0] = (__bf16)(v.x * r); o[1] = (__bf16)(v.y * r);
    o[2] = (__bf16)(v.z * r); o[3] = (__bf16)(v.w * r);
}

// ---------------------------------------------------------------------------
// Kernel 6: fused dual GEMM + bias + swish, pure-bf16 WMMA hot loop.
//   out = swish( meanb @ Wl + xb @ Wr + b )
// One wave -> two adjacent 16x16 N-tiles sharing the A fragment.
// Block = 8 waves covers 16 rows x 256 cols; grid = (N/16, 512/256).
// ---------------------------------------------------------------------------
__global__ __launch_bounds__(256)
void sage_gemm(const __bf16* __restrict__ meanb, const __bf16* __restrict__ xb,
               const __bf16* __restrict__ WTl, const __bf16* __restrict__ WTr,
               const float* __restrict__ bias, float* __restrict__ out) {
    const int wave = threadIdx.x >> 5;
    const int lane = threadIdx.x & 31;
    const int lm   = lane & 15;           // row (A) / col (B,C) within tile
    const int lh   = lane >> 4;           // half-wave selector
    const int m0   = blockIdx.x * 16;
    const int n0   = blockIdx.y * 256 + wave * 32;   // two tiles: n0, n0+16
    const int m    = m0 + lm;
    const int na   = n0 + lm;             // tile 0 column for this lane
    const int nb   = na + 16;             // tile 1 column for this lane

    v8f acc0 = {};
    v8f acc1 = {};

    #pragma unroll
    for (int pass = 0; pass < 2; ++pass) {
        const __bf16* A    = pass ? xb  : meanb;
        const __bf16* WT   = pass ? WTr : WTl;
        const __bf16* arow = A  + (size_t)m  * DDIM;
        const __bf16* wra  = WT + (size_t)na * DDIM;
        const __bf16* wrb  = WT + (size_t)nb * DDIM;

        for (int k = 0; k < DDIM; k += 32) {
            // A fragment: lane holds K = [k + lh*16, +16) contiguous
            const v8bf alo = *(const v8bf*)(arow + k + lh * 16);
            const v8bf ahi = *(const v8bf*)(arow + k + lh * 16 + 8);
            v16bf af;
            #pragma unroll
            for (int i = 0; i < 8; ++i) { af[i] = alo[i]; af[i + 8] = ahi[i]; }

            // B fragments: lane holds col n, K = [k + lh*16, +16) contiguous
            const v16bf bf0 = *(const v16bf*)(wra + k + lh * 16);
            const v16bf bf1 = *(const v16bf*)(wrb + k + lh * 16);

            acc0 = __builtin_amdgcn_wmma_f32_16x16x32_bf16(
                       false, af, false, bf0, (short)0, acc0, false, false);
            acc1 = __builtin_amdgcn_wmma_f32_16x16x32_bf16(
                       false, af, false, bf1, (short)0, acc1, false, false);
        }
    }

    // Epilogue: bias + swish.  C/D layout: lane -> col n, VGPR r -> row m0+lh*8+r
    const float bva = bias[na];
    const float bvb = bias[nb];
    float* orowa = out + (size_t)(m0 + lh * 8) * DDIM + na;
    float* orowb = out + (size_t)(m0 + lh * 8) * DDIM + nb;
    #pragma unroll
    for (int r = 0; r < 8; ++r) {
        float h0 = acc0[r] + bva;
        float h1 = acc1[r] + bvb;
        float s0 = 1.0f / (1.0f + __expf(-h0));
        float s1 = 1.0f / (1.0f + __expf(-h1));
        orowa[(size_t)r * DDIM] = h0 * s0;
        orowb[(size_t)r * DDIM] = h1 * s1;
    }
}

// ---------------------------------------------------------------------------
// Host-side launch
// ---------------------------------------------------------------------------
extern "C" void kernel_launch(void* const* d_in, const int* in_sizes, int n_in,
                              void* d_out, int out_size, void* d_ws, size_t ws_size,
                              hipStream_t stream) {
    const float* x   = (const float*)d_in[0];
    const int*   ei  = (const int*)d_in[1];
    const float* Wl  = (const float*)d_in[2];
    const float* Wr  = (const float*)d_in[3];
    const float* b   = (const float*)d_in[4];
    float*       out = (float*)d_out;

    const int N = in_sizes[0] / DDIM;     // 10000
    const int E = in_sizes[1] / 2;        // 160000
    const int* src = ei;
    const int* dst = ei + E;

    // Workspace layout (bytes, all chunks 32B-aligned):
    //   agg   f32   N*512*4   = 20,480,000
    //   deg   f32   N*4       =     40,000
    //   WTl   bf16  512*512*2 =    524,288
    //   WTr   bf16  512*512*2 =    524,288
    //   xb    bf16  N*512*2   = 10,240,000
    //   meanb bf16  N*512*2   = 10,240,000   (total ~42.05 MB)
    char*   ws    = (char*)d_ws;
    size_t  off   = 0;
    float*  agg   = (float*)(ws + off);  off += (size_t)N * DDIM * 4;
    float*  deg   = (float*)(ws + off);  off += (size_t)N * 4;
    off = (off + 31) & ~(size_t)31;
    __bf16* WTl   = (__bf16*)(ws + off); off += (size_t)DDIM * DDIM * 2;
    __bf16* WTr   = (__bf16*)(ws + off); off += (size_t)DDIM * DDIM * 2;
    __bf16* xb    = (__bf16*)(ws + off); off += (size_t)N * DDIM * 2;
    __bf16* meanb = (__bf16*)(ws + off);

    const int nfeat = N * DDIM;

    // 1) zero accumulators
    sage_zero<<<(nfeat + N + 255) / 256, 256, 0, stream>>>(agg, deg, nfeat, N);
    // 2) weight transpose + bf16 convert
    sage_wt<<<(DDIM * DDIM + 255) / 256, 256, 0, stream>>>(Wl, Wr, WTl, WTr);
    // 3) x -> bf16
    sage_xcvt<<<(nfeat / 4 + 255) / 256, 256, 0, stream>>>(x, xb, nfeat);
    // 4) edge scatter-add
    sage_scatter<<<E, 128, 0, stream>>>(x, src, dst, agg, deg, E);
    // 5) mean + bf16 convert
    sage_mean<<<(nfeat / 4 + 255) / 256, 256, 0, stream>>>(agg, deg, meanb, nfeat);
    // 6) fused dual-GEMM + bias + swish (WMMA)
    dim3 grid(N / 16, DDIM / 256);        // (625, 2), 8 waves/block, 2 tiles/wave
    sage_gemm<<<grid, 256, 0, stream>>>(meanb, xb, WTl, WTr, b, out);
}
